// GATLayerEdgeAverage_84765474554045
// MI455X (gfx1250) — matrix operand order repaired and checked by reading, not compiled, not run on hardware
//
#include <hip/hip_runtime.h>

// ---- vector typedefs ----
typedef __attribute__((ext_vector_type(16))) __bf16 v16bf;
typedef __attribute__((ext_vector_type(4)))  __bf16 v4bf;
typedef __attribute__((ext_vector_type(8)))  float  v8f;
typedef __attribute__((ext_vector_type(4)))  float  v4f;

#define D_IO 128              // d_i == d_o == 128
#define K_TOT 256             // 2*d_i
#define EPS_F 1e-6f
#define NFRAG 64              // 8 nt * 8 kt weight fragments
#define FRAG_BF16 (32 * 16)   // 32 lanes * 16 bf16 per fragment

// ---------------- zero (float4 per thread) ----------------
__global__ void gat_zero4(float* __restrict__ p, int n4) {
    int i = blockIdx.x * blockDim.x + threadIdx.x;
    if (i < n4) ((v4f*)p)[i] = (v4f){0.f, 0.f, 0.f, 0.f};
}

// ---------------- x (f32) -> xh (bf16) ----------------
__global__ void gat_cvt_x(const float* __restrict__ x, __bf16* __restrict__ xh, int n4) {
    int i = blockIdx.x * blockDim.x + threadIdx.x;
    if (i < n4) {
        v4f v = ((const v4f*)x)[i];
        v4bf o;
        o.x = (__bf16)v.x; o.y = (__bf16)v.y; o.z = (__bf16)v.z; o.w = (__bf16)v.w;
        ((v4bf*)xh)[i] = o;
    }
}

// ---- pack Wf [128][256] f32 into WMMA A-fragment order (bf16) ----
// frag f = nt*8 + kt ; per-lane 16 contiguous bf16:
//   lane<16 : row=nt*16+lane,      elems = Wf[row][kt*32 + 0..7],   Wf[row][kt*32 + 8..15]
//   lane>=16: row=nt*16+(lane-16), elems = Wf[row][kt*32 + 16..23], Wf[row][kt*32 + 24..31]
__global__ void gat_pack_w(const float* __restrict__ Wf, __bf16* __restrict__ apack) {
    int t = blockIdx.x * blockDim.x + threadIdx.x;
    if (t >= NFRAG * 32) return;
    int f = t >> 5, lane = t & 31;
    int nt = f >> 3, kt = f & 7;
    int half = lane >> 4;
    int row = nt * 16 + (lane & 15);
    int c0 = kt * 32 + (half ? 16 : 0);
    int c1 = kt * 32 + (half ? 24 : 8);
    __bf16* dst = apack + (size_t)t * 16;
#pragma unroll
    for (int j = 0; j < 8; ++j) dst[j]     = (__bf16)Wf[row * K_TOT + c0 + j];
#pragma unroll
    for (int j = 0; j < 8; ++j) dst[8 + j] = (__bf16)Wf[row * K_TOT + c1 + j];
}

// ------- main: weights staged in LDS, one wave32 per 16-edge tile, grid-stride -------
__global__ __launch_bounds__(256) void gat_edge_wmma(
    const __bf16* __restrict__ xh,     // [N][128] bf16
    const __bf16* __restrict__ apack,  // [64][32][16] bf16 packed Wf fragments
    const float*  __restrict__ bf,     // [128]
    const float*  __restrict__ ww,     // [256]
    const float*  __restrict__ bw,     // [1]
    const int*    __restrict__ src,
    const int*    __restrict__ tgt,
    float* __restrict__ out,           // [N][128] accumulators
    float* __restrict__ asum,          // [N]
    int E, int numTiles, int totalWaves)
{
    // 64 KB of loop-invariant packed weights in LDS (WGP has 320 KB)
    __shared__ __bf16 lw[NFRAG * FRAG_BF16];

    // cooperative one-time fill: 64 KB = 4096 uint4
    {
        const uint4* g = (const uint4*)apack;
        uint4* l = (uint4*)lw;
        const int n16 = NFRAG * FRAG_BF16 * (int)sizeof(__bf16) / 16;   // 4096
        for (int i = threadIdx.x; i < n16; i += blockDim.x) l[i] = g[i];
    }
    __syncthreads();

    const int lane = threadIdx.x & 31;
    const int half = lane >> 4;                   // 0: K lo 16, 1: K hi 16 of each 32-chunk
    const int choff = half * 8;                   // D layout: lane<16 -> M 0..7, lane>=16 -> M 8..15
    const int wave0 = blockIdx.x * (blockDim.x >> 5) + (threadIdx.x >> 5);
    const float bw0 = bw[0];

#pragma unroll 1
    for (int tile = wave0; tile < numTiles; tile += totalWaves) {
        // keep LDS reads inside the tile loop (defeat LICM -> scratch spill)
        asm volatile("" ::: "memory");

        const int eL   = tile * 16 + (lane & 15);
        const bool valid = (eL < E);
        const int e  = valid ? eL : (E - 1);      // clamp, keep EXEC full for WMMA
        const int si = src[e];
        const int ti = tgt[e];

        // ---- gather B fragments (edge rows, bf16) + fused attention dot ----
        v16bf bfrag[8];
        float adot = 0.f;
#pragma unroll
        for (int kt = 0; kt < 8; ++kt) {
            const __bf16* base = (kt < 4)
                ? (xh + (size_t)si * D_IO + kt * 32)
                : (xh + (size_t)ti * D_IO + (kt - 4) * 32);
            bfrag[kt] = *(const v16bf*)(base + half * 16);   // 32B contiguous per lane
            const float* wp = ww + kt * 32 + half * 16;
#pragma unroll
            for (int j = 0; j < 16; ++j)
                adot += (float)bfrag[kt][j] * wp[j];
        }
        // combine the two K-halves held by lane L and lane L+16
        float a_e = adot + __shfl_xor(adot, 16, 32) + bw0;

        // nt loop deliberately NOT unrolled: keeps the 8 LDS fragment loads per
        // iteration dependent on nt, so they cannot be hoisted out of the tile
        // loop (which previously caused a 512-VGPR hoist + scratch spill).
#pragma unroll 1
        for (int nt = 0; nt < 8; ++nt) {
            v8f acc = {0.f, 0.f, 0.f, 0.f, 0.f, 0.f, 0.f, 0.f};
#pragma unroll
            for (int kt = 0; kt < 8; ++kt) {
                const v16bf* ap = (const v16bf*)(lw + ((nt * 8 + kt) * 32 + lane) * 16);
                acc = __builtin_amdgcn_wmma_f32_16x16x32_bf16(
                    false, *ap, false, bfrag[kt], (short)0, acc, false, false);
            }
            if (valid) {
                const int cb = nt * 16 + choff;
                float* orow = out + (size_t)ti * D_IO + cb;
#pragma unroll
                for (int r = 0; r < 8; ++r) {
                    float y = acc[r] + bf[cb + r];
                    y = y > 0.f ? y : 0.f;
                    atomicAdd(orow + r, y * a_e);
                }
            }
        }
        if (valid && half == 0) atomicAdd(asum + ti, a_e);
    }
}

// ---------------- finalize: o /= (a_sum + eps) ----------------
__global__ void gat_fin(float* __restrict__ out, const float* __restrict__ asum, int n4) {
    int i = blockIdx.x * blockDim.x + threadIdx.x;
    if (i < n4) {
        float s = asum[i >> 5] + EPS_F;           // 32 float4's per 128-wide row
        v4f v = ((v4f*)out)[i];
        v.x /= s; v.y /= s; v.z /= s; v.w /= s;
        ((v4f*)out)[i] = v;
    }
}

extern "C" void kernel_launch(void* const* d_in, const int* in_sizes, int n_in,
                              void* d_out, int out_size, void* d_ws, size_t ws_size,
                              hipStream_t stream) {
    const float* x  = (const float*)d_in[0];   // [N,128]
    const float* Wf = (const float*)d_in[1];   // [128,256]
    const float* bf = (const float*)d_in[2];   // [128]
    const float* Ww = (const float*)d_in[3];   // [1,256]
    const float* bw = (const float*)d_in[4];   // [1]
    const int*  src = (const int*)d_in[5];     // [E]
    const int*  tgt = (const int*)d_in[6];     // [E]

    const int N = in_sizes[0] / D_IO;
    const int E = in_sizes[5];

    // ---- workspace layout ----
    char* ws = (char*)d_ws;
    size_t off = 0;
    __bf16* xh = (__bf16*)(ws + off);
    off += (size_t)N * D_IO * sizeof(__bf16);
    off = (off + 255) & ~(size_t)255;
    __bf16* apack = (__bf16*)(ws + off);
    off += (size_t)NFRAG * FRAG_BF16 * sizeof(__bf16);
    off = (off + 255) & ~(size_t)255;
    float* asum = (float*)(ws + off);

    float* out = (float*)d_out;

    const int TPB = 256;
    // zero output accumulators + a_sum
    {
        int n4 = (N * D_IO) / 4;
        gat_zero4<<<(n4 + TPB - 1) / TPB, TPB, 0, stream>>>(out, n4);
        int n4a = (N + 3) / 4;
        gat_zero4<<<(n4a + TPB - 1) / TPB, TPB, 0, stream>>>(asum, n4a);
    }
    // convert x -> bf16
    {
        int n4 = (N * D_IO) / 4;
        gat_cvt_x<<<(n4 + TPB - 1) / TPB, TPB, 0, stream>>>(x, xh, n4);
    }
    // pack weights into WMMA fragment order
    gat_pack_w<<<(NFRAG * 32 + TPB - 1) / TPB, TPB, 0, stream>>>(Wf, apack);

    // main edge kernel: weights in LDS, grid-stride over 16-edge tiles
    {
        int numTiles = (E + 15) / 16;
        int wavesPerBlk = TPB / 32;
        int maxBlocks = 1024;
        int needBlocks = (numTiles + wavesPerBlk - 1) / wavesPerBlk;
        int blocks = needBlocks < maxBlocks ? needBlocks : maxBlocks;
        int totalWaves = blocks * wavesPerBlk;
        gat_edge_wmma<<<blocks, TPB, 0, stream>>>(xh, apack, bf, Ww, bw, src, tgt,
                                                  out, asum, E, numTiles, totalWaves);
    }
    // finalize divide
    {
        int n4 = (N * D_IO) / 4;
        gat_fin<<<(n4 + TPB - 1) / TPB, TPB, 0, stream>>>(out, asum, n4);
    }
}
